// DecoderBlock_31834297598399
// MI455X (gfx1250) — matrix operand order repaired and checked by reading, not compiled, not run on hardware
//
#include <hip/hip_runtime.h>
#include <hip/hip_bf16.h>
#include <math.h>

typedef __bf16 bf16_t;
typedef __attribute__((ext_vector_type(2)))  __bf16 bf16x2;
typedef __attribute__((ext_vector_type(16))) __bf16 bf16x16;
typedef __attribute__((ext_vector_type(8)))  float  f32x8;

// ---------------------------------------------------------------------------
// bf16-WMMA GEMM:  C[z] = epi( A[z](MxK) * B[z](KxN) )
//   REQUIRES: M % 128 == 0, N % 64 == 0, K % 32 == 0, buffers < 2^32 bytes
//   epi: +bias[n], exact GELU, *rowscale[m*rs_stride], +resid[m][n], +=C
// Block tile 128x64, 256 threads = 8 waves in 4x2; each wave computes 32x32
// with four 16x16x32 WMMAs per K-step. LDS tiles are stored fragment-major
// (contiguous k: lane half h holds k = 16h..16h+15) so each fragment load is 2x ds_load_b128. Global
// addresses are byte offsets (saddr + zext(voffset) form) bumped per K-step.
// TRANSB is a template parameter so the k-loop is branch-free.
// ---------------------------------------------------------------------------
template<int TRANSB>
__device__ __forceinline__
void gemm_body(const float* __restrict__ A, int lda,
               const float* __restrict__ B, int ldb,
               float* __restrict__ C, int ldc,
               const float* __restrict__ bias,
               const float* __restrict__ resid, int ldr,
               const float* __restrict__ rowscale, int rs_stride,
               int K, int actGelu, int accum,
               int mrow0, int ncol0,
               bf16_t (*Afr)[32][16], bf16_t (*Bfr)[32][16])
{
    const int tid  = threadIdx.x;
    const int wid  = tid >> 5;
    const int lane = tid & 31;
    const int wr   = wid >> 1;   // 0..3 : 32-row band
    const int wc   = wid & 1;    // 0..1 : 32-col band

    // --- staging constants (per-thread invariant) ---
    const int a_kp   = tid & 15;                 // A: k-pair index (k = 2*kp)
    const int a_mm   = tid >> 4;                 // A: row within band
    const int a_lane = a_mm + 16 * (a_kp >> 3);
    const int a_idx  = 2 * (a_kp & 7);
    const int bt_lane = (tid >> 4) + 16 * ((tid & 15) >> 3);
    const int bt_idx  = 2 * (tid & 7);
    const int bn_n  = tid & 63;                  // normal-B: contiguous n per lane
    const int bn_f  = bn_n >> 4;
    const int bn_nl = bn_n & 15;

    // --- strength-reduced BYTE offsets over uniform bases ---
    const char* Ab = (const char*)A;
    const char* Bb = (const char*)B;
    unsigned aoff[8];
    #pragma unroll
    for (int i = 0; i < 8; ++i)
        aoff[i] = ((unsigned)(mrow0 + i * 16 + a_mm) * (unsigned)lda + 2u * a_kp) * 4u;
    unsigned boff[4];
    unsigned bstep;
    if (TRANSB) {
        #pragma unroll
        for (int i = 0; i < 4; ++i)
            boff[i] = ((unsigned)(ncol0 + i * 16 + (tid >> 4)) * (unsigned)ldb + 2u * (tid & 15)) * 4u;
        bstep = 128u;
    } else {
        #pragma unroll
        for (int i = 0; i < 4; ++i)
            boff[i] = ((unsigned)(2 * ((tid >> 6) + 4 * i)) * (unsigned)ldb + (unsigned)(ncol0 + bn_n)) * 4u;
        bstep = 128u * (unsigned)ldb;
    }
    const unsigned ldb_bytes = 4u * (unsigned)ldb;

    f32x8 acc[2][2] = {};

    for (int k0 = 0; k0 < K; k0 += 32) {
        // ---- unconditional global loads (batched, overlap prior WMMAs) ----
        float2 av[8];
        #pragma unroll
        for (int i = 0; i < 8; ++i) {
            av[i] = *(const float2*)(Ab + aoff[i]);
            aoff[i] += 128u;
        }
        float2 bv[4];
        if (TRANSB) {
            #pragma unroll
            for (int i = 0; i < 4; ++i) {
                bv[i] = *(const float2*)(Bb + boff[i]);
                boff[i] += bstep;
            }
        } else {
            #pragma unroll
            for (int i = 0; i < 4; ++i) {
                bv[i].x = *(const float*)(Bb + boff[i]);
                bv[i].y = *(const float*)(Bb + boff[i] + ldb_bytes);
                boff[i] += bstep;
            }
        }

        __syncthreads();   // previous iteration done reading LDS

        // ---- pack fp32 pairs -> bf16x2, scatter into fragment layout ----
        #pragma unroll
        for (int i = 0; i < 8; ++i) {
            *(bf16x2*)&Afr[i][a_lane][a_idx] = bf16x2{(bf16_t)av[i].x, (bf16_t)av[i].y};
        }
        if (TRANSB) {
            #pragma unroll
            for (int i = 0; i < 4; ++i) {
                *(bf16x2*)&Bfr[i][bt_lane][bt_idx] = bf16x2{(bf16_t)bv[i].x, (bf16_t)bv[i].y};
            }
        } else {
            #pragma unroll
            for (int i = 0; i < 4; ++i) {
                int kp = (tid >> 6) + 4 * i;
                *(bf16x2*)&Bfr[bn_f][bn_nl + 16 * (kp >> 3)][2 * (kp & 7)] =
                    bf16x2{(bf16_t)bv[i].x, (bf16_t)bv[i].y};
            }
        }

        __syncthreads();

        // ---- fragment loads: 2x ds_load_b128 each ----
        const bf16x16 a0 = *(const bf16x16*)&Afr[wr * 2    ][lane][0];
        const bf16x16 a1 = *(const bf16x16*)&Afr[wr * 2 + 1][lane][0];
        const bf16x16 b0 = *(const bf16x16*)&Bfr[wc * 2    ][lane][0];
        const bf16x16 b1 = *(const bf16x16*)&Bfr[wc * 2 + 1][lane][0];

        acc[0][0] = __builtin_amdgcn_wmma_f32_16x16x32_bf16(false, a0, false, b0, (short)0, acc[0][0], false, false);
        acc[0][1] = __builtin_amdgcn_wmma_f32_16x16x32_bf16(false, a0, false, b1, (short)0, acc[0][1], false, false);
        acc[1][0] = __builtin_amdgcn_wmma_f32_16x16x32_bf16(false, a1, false, b0, (short)0, acc[1][0], false, false);
        acc[1][1] = __builtin_amdgcn_wmma_f32_16x16x32_bf16(false, a1, false, b1, (short)0, acc[1][1], false, false);
    }

    // ---- epilogue: C/D layout lane%16 = N, lane/16 picks 8-row band ----
    const int nlo = lane & 15;
    const int mb  = (lane >> 4) * 8;
    #pragma unroll
    for (int s = 0; s < 2; ++s) {
        #pragma unroll
        for (int t = 0; t < 2; ++t) {
            int n = ncol0 + wc * 32 + t * 16 + nlo;
            float bvl = bias ? bias[n] : 0.0f;
            #pragma unroll
            for (int r = 0; r < 8; ++r) {
                int m = mrow0 + wr * 32 + s * 16 + mb + r;
                float v = acc[s][t][r] + bvl;
                if (actGelu) v = 0.5f * v * (1.0f + erff(v * 0.70710678118654752f));
                if (rowscale) v *= rowscale[(long)m * rs_stride];
                if (resid) v += resid[(long)m * ldr + n];
                if (accum) v += C[(long)m * ldc + n];
                C[(long)m * ldc + n] = v;
            }
        }
    }
}

__global__ __launch_bounds__(256)
void wmma_gemm_kernel(const float* __restrict__ A, int lda, long zsa,
                      const float* __restrict__ B, int ldb, long zsb,
                      float* __restrict__ C, int ldc, long zsc,
                      const float* __restrict__ bias,
                      const float* __restrict__ resid, int ldr,
                      const float* __restrict__ rowscale, int rs_stride,
                      int K, int transB, int actGelu, int accum)
{
    // fragment-major LDS: [16-row band][lane][16 bf16]  (32B per lane, 16B aligned)
    __shared__ __align__(32) bf16_t Afr[8][32][16];   // 128 rows x 32 k
    __shared__ __align__(32) bf16_t Bfr[4][32][16];   // 32 k x 64 n

    const int z     = blockIdx.z;
    const int mrow0 = blockIdx.y * 128;
    const int ncol0 = blockIdx.x * 64;
    A += (long)z * zsa;
    B += (long)z * zsb;
    C += (long)z * zsc;

    if (transB)
        gemm_body<1>(A, lda, B, ldb, C, ldc, bias, resid, ldr, rowscale, rs_stride,
                     K, actGelu, accum, mrow0, ncol0, Afr, Bfr);
    else
        gemm_body<0>(A, lda, B, ldb, C, ldc, bias, resid, ldr, rowscale, rs_stride,
                     K, actGelu, accum, mrow0, ncol0, Afr, Bfr);
}

// ---------------------------------------------------------------------------
// tiny linear for the adaLN embedding: out[b][n] = sin[b]·W[:,n] + bias[n]
// (M=4, N=2048, K=1024 -> ~8 MFLOP, VALU is the right tool)
// ---------------------------------------------------------------------------
__global__ __launch_bounds__(256)
void ada_linear_kernel(const float* __restrict__ sinb, const float* __restrict__ W,
                       const float* __restrict__ bias, float* __restrict__ out)
{
    int n = blockIdx.x * 256 + threadIdx.x;   // 0..2047
    int b = blockIdx.y;                        // 0..3
    const float* xr = sinb + b * 1024;
    const float* wc = W + n;
    float s = bias[n];
    #pragma unroll 8
    for (int k = 0; k < 1024; ++k) s += xr[k] * wc[(long)k * 2048];
    out[b * 2048 + n] = s;
}

// ---------------------------------------------------------------------------
// sinusoidal(t) + SiLU  -> out[B][1024]
// ---------------------------------------------------------------------------
__global__ void sin_embed_kernel(const int* __restrict__ ts, float* __restrict__ out)
{
    int b = blockIdx.x;
    int d = threadIdx.x;                        // 1024 threads
    float t = (float)ts[b];
    int i = (d < 512) ? d : d - 512;
    float freq = __expf((float)i * (-logf(10000.0f) / 511.0f));
    float ang  = t * freq;
    float v    = (d < 512) ? __sinf(ang) : __cosf(ang);
    v = v / (1.0f + __expf(-v));                // SiLU
    out[b * 1024 + d] = v;
}

// ---------------------------------------------------------------------------
// adaLN: out = xn * (1 + scale_b) + shift_b ; emb[b] = [scale(1024)|shift(1024)]
// ---------------------------------------------------------------------------
__global__ __launch_bounds__(256)
void ada_ln_kernel(const float* __restrict__ x, const float* __restrict__ emb,
                   float* __restrict__ out)
{
    __shared__ float red[256];
    const long row = blockIdx.x;
    const int tid = threadIdx.x;
    const float* xr = x + row * 1024;
    float lv[4]; float s = 0.0f;
    #pragma unroll
    for (int i = 0; i < 4; ++i) { lv[i] = xr[tid + i * 256]; s += lv[i]; }
    red[tid] = s; __syncthreads();
    for (int o = 128; o > 0; o >>= 1) { if (tid < o) red[tid] += red[tid + o]; __syncthreads(); }
    float mean = red[0] * (1.0f / 1024.0f); __syncthreads();
    s = 0.0f;
    #pragma unroll
    for (int i = 0; i < 4; ++i) { float d = lv[i] - mean; s += d * d; }
    red[tid] = s; __syncthreads();
    for (int o = 128; o > 0; o >>= 1) { if (tid < o) red[tid] += red[tid + o]; __syncthreads(); }
    float rstd = rsqrtf(red[0] * (1.0f / 1024.0f) + 1e-5f);
    const float* e = emb + (row >> 10) * 2048;
    #pragma unroll
    for (int i = 0; i < 4; ++i) {
        int d = tid + i * 256;
        out[row * 1024 + d] = (lv[i] - mean) * rstd * (1.0f + e[d]) + e[1024 + d];
    }
}

// ---------------------------------------------------------------------------
// LayerNorm with g/b; optional residual ADDED AFTER LN (for final output)
// ---------------------------------------------------------------------------
__global__ __launch_bounds__(256)
void layernorm_kernel(const float* __restrict__ x, const float* __restrict__ g,
                      const float* __restrict__ b, const float* __restrict__ post_resid,
                      float* __restrict__ out)
{
    __shared__ float red[256];
    const long row = blockIdx.x;
    const int tid = threadIdx.x;
    const float* xr = x + row * 1024;
    float lv[4]; float s = 0.0f;
    #pragma unroll
    for (int i = 0; i < 4; ++i) { lv[i] = xr[tid + i * 256]; s += lv[i]; }
    red[tid] = s; __syncthreads();
    for (int o = 128; o > 0; o >>= 1) { if (tid < o) red[tid] += red[tid + o]; __syncthreads(); }
    float mean = red[0] * (1.0f / 1024.0f); __syncthreads();
    s = 0.0f;
    #pragma unroll
    for (int i = 0; i < 4; ++i) { float d = lv[i] - mean; s += d * d; }
    red[tid] = s; __syncthreads();
    for (int o = 128; o > 0; o >>= 1) { if (tid < o) red[tid] += red[tid + o]; __syncthreads(); }
    float rstd = rsqrtf(red[0] * (1.0f / 1024.0f) + 1e-5f);
    #pragma unroll
    for (int i = 0; i < 4; ++i) {
        int d = tid + i * 256;
        float v = (lv[i] - mean) * rstd * g[d] + b[d];
        if (post_resid) v += post_resid[row * 1024 + d];
        out[row * 1024 + d] = v;
    }
}

// ---------------------------------------------------------------------------
// q = rmsnorm(q, g) + q  (in place)
// ---------------------------------------------------------------------------
__global__ __launch_bounds__(256)
void rmsnorm_addself_kernel(float* __restrict__ q, const float* __restrict__ g)
{
    __shared__ float red[256];
    const long row = blockIdx.x;
    const int tid = threadIdx.x;
    float* p = q + row * 1024;
    float lv[4]; float s = 0.0f;
    #pragma unroll
    for (int i = 0; i < 4; ++i) { lv[i] = p[tid + i * 256]; s += lv[i] * lv[i]; }
    red[tid] = s; __syncthreads();
    for (int o = 128; o > 0; o >>= 1) { if (tid < o) red[tid] += red[tid + o]; __syncthreads(); }
    float rr = rsqrtf(red[0] * (1.0f / 1024.0f) + 1e-6f);
    #pragma unroll
    for (int i = 0; i < 4; ++i) {
        int d = tid + i * 256;
        p[d] = lv[i] * rr * g[d] + lv[i];
    }
}

// ---------------------------------------------------------------------------
// row softmax of exp(alpha*(s - max)) / sum, in place
// ---------------------------------------------------------------------------
__global__ __launch_bounds__(256)
void softmax_kernel(float* __restrict__ s, int L, float alpha)
{
    __shared__ float red[256];
    float* p = s + (long)blockIdx.x * L;
    const int tid = threadIdx.x;
    float m = -3.402823466e38f;
    for (int i = tid; i < L; i += 256) m = fmaxf(m, p[i]);
    red[tid] = m; __syncthreads();
    for (int o = 128; o > 0; o >>= 1) { if (tid < o) red[tid] = fmaxf(red[tid], red[tid + o]); __syncthreads(); }
    m = red[0]; __syncthreads();
    float sum = 0.0f;
    for (int i = tid; i < L; i += 256) { float e = __expf((p[i] - m) * alpha); p[i] = e; sum += e; }
    red[tid] = sum; __syncthreads();
    for (int o = 128; o > 0; o >>= 1) { if (tid < o) red[tid] += red[tid + o]; __syncthreads(); }
    float inv = 1.0f / red[0];
    for (int i = tid; i < L; i += 256) p[i] *= inv;
}

// ---------------------------------------------------------------------------
// gate logits + top-2 softmax -> combine[rows][4]
// ---------------------------------------------------------------------------
__global__ void gate_topk_kernel(const float* __restrict__ xln,
                                 const float* __restrict__ gw,
                                 const float* __restrict__ gb,
                                 float* __restrict__ combine, int rows)
{
    int t = blockIdx.x * blockDim.x + threadIdx.x;
    if (t >= rows) return;
    const float* xr = xln + (long)t * 1024;
    float lg[4] = {gb[0], gb[1], gb[2], gb[3]};
    for (int k = 0; k < 1024; ++k) {
        float xv = xr[k];
        const float* w = gw + k * 4;
        #pragma unroll
        for (int e = 0; e < 4; ++e) lg[e] += xv * w[e];
    }
    int i0 = 0;
    #pragma unroll
    for (int e = 1; e < 4; ++e) if (lg[e] > lg[i0]) i0 = e;
    int i1 = -1;
    #pragma unroll
    for (int e = 0; e < 4; ++e) { if (e == i0) continue; if (i1 < 0 || lg[e] > lg[i1]) i1 = e; }
    float g1 = __expf(lg[i1] - lg[i0]);
    float zi = 1.0f / (1.0f + g1);
    #pragma unroll
    for (int e = 0; e < 4; ++e) combine[(long)t * 4 + e] = 0.0f;
    combine[(long)t * 4 + i0] = zi;
    combine[(long)t * 4 + i1] = g1 * zi;
}

// ---------------------------------------------------------------------------
extern "C" void kernel_launch(void* const* d_in, const int* in_sizes, int n_in,
                              void* d_out, int out_size, void* d_ws, size_t ws_size,
                              hipStream_t stream)
{
    (void)in_sizes; (void)n_in; (void)out_size; (void)ws_size;
    const float* x     = (const float*)d_in[0];
    const float* enc   = (const float*)d_in[1];
    const int*   ts    = (const int*)d_in[2];
    const float* ada1w = (const float*)d_in[3];
    const float* ada1b = (const float*)d_in[4];
    const float* ada2w = (const float*)d_in[5];
    const float* ada2b = (const float*)d_in[6];
    const float* ln3g  = (const float*)d_in[7];
    const float* ln3b  = (const float*)d_in[8];
    const float* a1qw = (const float*)d_in[9],  *a1qb = (const float*)d_in[10];
    const float* a1kw = (const float*)d_in[11], *a1kb = (const float*)d_in[12];
    const float* a1vw = (const float*)d_in[13], *a1vb = (const float*)d_in[14];
    const float* a1pw = (const float*)d_in[15], *a1pb = (const float*)d_in[16];
    const float* a2qw = (const float*)d_in[17], *a2qb = (const float*)d_in[18];
    const float* a2kw = (const float*)d_in[19], *a2kb = (const float*)d_in[20];
    const float* a2vw = (const float*)d_in[21], *a2vb = (const float*)d_in[22];
    const float* a2pw = (const float*)d_in[23], *a2pb = (const float*)d_in[24];
    const float* qnrm = (const float*)d_in[25];
    const float* knrm = (const float*)d_in[26];
    const float* gatw = (const float*)d_in[27];
    const float* gatb = (const float*)d_in[28];
    const float* w1   = (const float*)d_in[29];
    const float* b1   = (const float*)d_in[30];
    const float* w2   = (const float*)d_in[31];
    const float* b2   = (const float*)d_in[32];
    const float* moeg = (const float*)d_in[33];
    const float* moeb = (const float*)d_in[34];

    const int M = 4096;            // B*T tokens
    const long SL = 1024L * 1024L; // per-(b,h) score slice / per-batch token slice

    // workspace layout (floats), hand-aliased across phases
    float* ws      = (float*)d_ws;
    float* sinbuf  = ws;                        // 4*1024
    float* emb     = sinbuf + 4096;             // 4*2048
    float* hbuf    = emb + 8192;                // 4096*1024  (later: xb)
    float* qbuf    = hbuf + (long)M * 1024;     // 4096*1024  (later: xln)
    float* kbuf    = qbuf + (long)M * 1024;     // 4096*1024  (later: moe acc)
    float* vbuf    = kbuf + (long)M * 1024;     // 4096*1024
    float* attn    = vbuf + (long)M * 1024;     // 4096*1024
    float* xa      = attn + (long)M * 1024;     // 4096*1024
    float* scores  = xa + (long)M * 1024;       // 16*1024*1024 (later: moe hidden)
    float* combine = scores + 16L * SL;         // 4096*4

    auto gemm = [&](const float* A, int lda, long zsa,
                    const float* B, int ldb, long zsb,
                    float* C, int ldc, long zsc,
                    const float* bias, const float* resid, int ldr,
                    const float* rsc, int rstride,
                    int Mm, int Nn, int Kk, int tb, int act, int acc, int Z) {
        dim3 g(Nn / 64, Mm / 128, Z);
        wmma_gemm_kernel<<<g, dim3(256), 0, stream>>>(
            A, lda, zsa, B, ldb, zsb, C, ldc, zsc,
            bias, resid, ldr, rsc, rstride, Kk, tb, act, acc);
    };

    auto sdpa = [&](const float* q, const float* k, const float* v, float* o) {
        for (int b = 0; b < 4; ++b) {
            const long boff = (long)b * SL;
            // scores[z] = q_head(z) * k_head(z)^T   (z = head, 16 per batch)
            gemm(q + boff, 1024, 64, k + boff, 1024, 64, scores, 1024, SL,
                 nullptr, nullptr, 0, nullptr, 0, 1024, 1024, 64, 1, 0, 0, 16);
            softmax_kernel<<<16 * 1024, 256, 0, stream>>>(scores, 1024, 0.125f);
            // out_head(z) = softmax * v_head(z)
            gemm(scores, 1024, SL, v + boff, 1024, 64, o + boff, 1024, 64,
                 nullptr, nullptr, 0, nullptr, 0, 1024, 64, 1024, 0, 0, 0, 16);
        }
    };

    // ---------------- self attention ----------------
    sin_embed_kernel<<<4, 1024, 0, stream>>>(ts, sinbuf);
    ada_linear_kernel<<<dim3(8, 4), 256, 0, stream>>>(sinbuf, ada1w, ada1b, emb);
    ada_ln_kernel<<<M, 256, 0, stream>>>(x, emb, hbuf);

    gemm(hbuf, 1024, 0, a1qw, 1024, 0, qbuf, 1024, 0, a1qb, nullptr, 0, nullptr, 0, M, 1024, 1024, 0, 0, 0, 1);
    gemm(hbuf, 1024, 0, a1kw, 1024, 0, kbuf, 1024, 0, a1kb, nullptr, 0, nullptr, 0, M, 1024, 1024, 0, 0, 0, 1);
    gemm(hbuf, 1024, 0, a1vw, 1024, 0, vbuf, 1024, 0, a1vb, nullptr, 0, nullptr, 0, M, 1024, 1024, 0, 0, 0, 1);

    sdpa(qbuf, kbuf, vbuf, attn);

    // xa = x + attn * Wp + bp
    gemm(attn, 1024, 0, a1pw, 1024, 0, xa, 1024, 0, a1pb, x, 1024, nullptr, 0, M, 1024, 1024, 0, 0, 0, 1);

    // ---------------- cross attention ----------------
    ada_linear_kernel<<<dim3(8, 4), 256, 0, stream>>>(sinbuf, ada2w, ada2b, emb);
    ada_ln_kernel<<<M, 256, 0, stream>>>(xa, emb, hbuf);

    gemm(hbuf, 1024, 0, a2qw, 1024, 0, qbuf, 1024, 0, a2qb, nullptr, 0, nullptr, 0, M, 1024, 1024, 0, 0, 0, 1);
    gemm(enc,  1024, 0, a2kw, 1024, 0, kbuf, 1024, 0, a2kb, nullptr, 0, nullptr, 0, M, 1024, 1024, 0, 0, 0, 1);
    gemm(enc,  1024, 0, a2vw, 1024, 0, vbuf, 1024, 0, a2vb, nullptr, 0, nullptr, 0, M, 1024, 1024, 0, 0, 0, 1);

    rmsnorm_addself_kernel<<<M, 256, 0, stream>>>(qbuf, qnrm);
    rmsnorm_addself_kernel<<<M, 256, 0, stream>>>(kbuf, knrm);

    sdpa(qbuf, kbuf, vbuf, attn);

    // xb (alias hbuf) = xa + attn * Wp2 + bp2
    gemm(attn, 1024, 0, a2pw, 1024, 0, hbuf, 1024, 0, a2pb, xa, 1024, nullptr, 0, M, 1024, 1024, 0, 0, 0, 1);

    // ---------------- MoE FFN ----------------
    // xln (alias qbuf) = LN3(xb)
    layernorm_kernel<<<M, 256, 0, stream>>>(hbuf, ln3g, ln3b, nullptr, qbuf);
    gate_topk_kernel<<<16, 256, 0, stream>>>(qbuf, gatw, gatb, combine, M);

    for (int e = 0; e < 4; ++e) {
        const float* w1e = w1 + (long)e * 1024 * 4096;
        const float* b1e = b1 + (long)e * 4096;
        const float* w2e = w2 + (long)e * 4096 * 1024;
        const float* b2e = b2 + (long)e * 1024;
        // he (alias scores) = gelu(xln * w1e + b1e)
        gemm(qbuf, 1024, 0, w1e, 4096, 0, scores, 4096, 0, b1e,
             nullptr, 0, nullptr, 0, M, 4096, 1024, 0, 1, 0, 1);
        // acc (alias kbuf): e==0 -> xln + c0*(he*w2e+b2e); else += ce*(...)
        gemm(scores, 4096, 0, w2e, 1024, 0, kbuf, 1024, 0, b2e,
             (e == 0 ? qbuf : nullptr), 1024, combine + e, 4,
             M, 1024, 4096, 0, 0, (e == 0 ? 0 : 1), 1);
    }

    // out = xb + LN_moe(acc)
    layernorm_kernel<<<M, 256, 0, stream>>>(kbuf, moeg, moeb, hbuf, (float*)d_out);
}
